// EXL2Linear_25039659335797
// MI455X (gfx1250) — matrix operand order repaired and checked by reading, not compiled, not run on hardware
//
#include <hip/hip_runtime.h>
#include <cstdint>

// ---------------------------------------------------------------------------
// EXL2 fused dequant-GEMM for MI455X (gfx1250, wave32, WMMA).
//   out[M,N] = input[M,K] @ W_eff,  W_eff[j,:] = dequant(W)[q_invperm[j],:]
//
// Preferred path (ws big enough): 3 phases
//   1) expand 4-bit scale codes -> fp32 [G,N] table        (1.5 MB ws)
//   2) dequant + row-permute W once -> W_eff^T fp16 [N,K]  (96 MB ws)
//      (n-major layout == WMMA-B-fragment-friendly; one-time ~5us HBM cost)
//   3) clean double-buffered async-LDS f16 WMMA GEMM (no VALU in inner loop)
// Fallback path (small ws): fused gather-dequant GEMM (round-2 kernel).
// ---------------------------------------------------------------------------

typedef _Float16 v16h __attribute__((ext_vector_type(16)));
typedef _Float16 v8h  __attribute__((ext_vector_type(8)));
typedef float    v8f  __attribute__((ext_vector_type(8)));
typedef int      v4i  __attribute__((ext_vector_type(4)));

#define MDIM 4096
#define KDIM 4096
#define NDIM 12288
#define GROUP 128
#define GDIM (KDIM / GROUP)

#define BM 128
#define BN 128
#define BK 32
#define NKT (KDIM / BK)

// ---- CDNA5 async global->LDS path (guarded so a name mismatch still builds)
#if __has_builtin(__builtin_amdgcn_global_load_async_to_lds_b128)
#define USE_ASYNC_LDS 1
#endif

__device__ __forceinline__ void async_copy_b128(const _Float16* g, _Float16* l) {
#ifdef USE_ASYNC_LDS
  __builtin_amdgcn_global_load_async_to_lds_b128(
      (__attribute__((address_space(1))) v4i*)(uintptr_t)g,
      (__attribute__((address_space(3))) v4i*)(uintptr_t)l,
      /*offset=*/0, /*cpol=*/0);
#else
  *(v8h*)l = *(const v8h*)g;   // sync fallback: vmem load + ds write
#endif
}

__device__ __forceinline__ void wait_async_lds() {
#ifdef USE_ASYNC_LDS
#if __has_builtin(__builtin_amdgcn_s_wait_asynccnt)
  __builtin_amdgcn_s_wait_asynccnt(0);
#else
  asm volatile("s_wait_asynccnt 0" ::: "memory");
#endif
#endif
}

// ---------------------------------------------------------------------------
// Phase 1: expand 4-bit scale codes -> fp32 [G, N] table. s=(code+1)^2*smax[g]
// ---------------------------------------------------------------------------
__global__ __launch_bounds__(256) void exl2_scales_kernel(
    const int* __restrict__ q_scale, const _Float16* __restrict__ q_scale_max,
    float* __restrict__ scales) {
  int idx = blockIdx.x * 256 + threadIdx.x;
  if (idx >= GDIM * NDIM) return;
  int g = idx / NDIM;
  int n = idx - g * NDIM;
  int packed = q_scale[g * (NDIM / 8) + (n >> 3)];
  int code = (packed >> ((n & 7) * 4)) & 0xF;
  float c1 = (float)(code + 1);
  scales[idx] = c1 * c1 * (float)q_scale_max[g];
}

// ---------------------------------------------------------------------------
// Phase 2: one-time dequant + row-permute, stored TRANSPOSED (n-major):
//   wt[n*K + j] = (nib(qw[ks>>3][n], ks&7) - 8) * scale(ks>>7, n), ks=invperm[j]
// Thread: fixed n, 8 consecutive j -> one b128 store; stores fully coalesced.
// ---------------------------------------------------------------------------
__global__ __launch_bounds__(256) void exl2_build_wt_kernel(
    const int* __restrict__ qw, const float* __restrict__ scales,
    const int* __restrict__ invperm, _Float16* __restrict__ wt) {
  int idx = blockIdx.x * 256 + threadIdx.x;   // over N * (K/8)
  int n = idx / (KDIM / 8);
  int jb = (idx - n * (KDIM / 8)) * 8;
  if (n >= NDIM) return;
  _Float16 tmp[8];
#pragma unroll
  for (int i = 0; i < 8; ++i) {
    int ks = invperm[jb + i];
    int w = qw[(size_t)(ks >> 3) * NDIM + n];
    float sc = scales[(size_t)(ks >> 7) * NDIM + n];
    int nib = (w >> ((ks & 7) * 4)) & 0xF;
    tmp[i] = (_Float16)((float)(nib - 8) * sc);
  }
  *(v8h*)(wt + (size_t)n * KDIM + jb) = *(v8h*)&tmp[0];
}

// ---------------------------------------------------------------------------
// Phase 3: clean f16 WMMA GEMM, out = x @ wt^T (wt is [N,K] n-major).
// Block: 256 threads = 8 waves; tile 128x128; K-step 32; wave = 64x32 sub-tile.
// Inner loop: 4 async b128 copies + 12 ds_load_b128 + 8 v_wmma per K-step.
// ---------------------------------------------------------------------------
__global__ __launch_bounds__(256) void exl2_gemm_pre_kernel(
    const _Float16* __restrict__ x, const _Float16* __restrict__ wt,
    const _Float16* __restrict__ bias, float* __restrict__ out) {
  __shared__ __align__(16) _Float16 Ab[2][BM * BK];  // [m][k]
  __shared__ __align__(16) _Float16 Bb[2][BN * BK];  // [n][k]

  const int tid = threadIdx.x;
  const int m0 = blockIdx.y * BM;
  const int n0 = blockIdx.x * BN;

  const int lane = tid & 31;
  const int wave = tid >> 5;
  const int wm = (wave >> 2) * 64;
  const int wn = (wave & 3) * 32;
  const int ml = lane & 15;
  const int kh = lane >> 4;

  v8f acc[4][2] = {};

  auto load_stage = [&](int st, int kt) {
    // A and B tiles are the same shape: 128 rows x 32 halves = 512 x 16B.
#pragma unroll
    for (int i = 0; i < 2; ++i) {
      int lin = tid + i * 256;
      int row = lin >> 2;
      int chunk = lin & 3;
      async_copy_b128(x + (size_t)(m0 + row) * KDIM + kt * BK + chunk * 8,
                      &Ab[st][row * BK + chunk * 8]);
      async_copy_b128(wt + (size_t)(n0 + row) * KDIM + kt * BK + chunk * 8,
                      &Bb[st][row * BK + chunk * 8]);
    }
  };

  auto compute_stage = [&](int st) {
    v16h afrag[4], bfrag[2];
#pragma unroll
    for (int mi = 0; mi < 4; ++mi) {
      const _Float16* base = &Ab[st][(wm + mi * 16 + ml) * BK + kh * 8];
      v8h lo = *(const v8h*)base;
      v8h hi = *(const v8h*)(base + 16);
      afrag[mi] = __builtin_shufflevector(lo, hi, 0, 1, 2, 3, 4, 5, 6, 7, 8, 9,
                                          10, 11, 12, 13, 14, 15);
    }
#pragma unroll
    for (int ni = 0; ni < 2; ++ni) {
      const _Float16* base = &Bb[st][(wn + ni * 16 + ml) * BK + kh * 8];
      v8h lo = *(const v8h*)base;
      v8h hi = *(const v8h*)(base + 16);
      bfrag[ni] = __builtin_shufflevector(lo, hi, 0, 1, 2, 3, 4, 5, 6, 7, 8, 9,
                                          10, 11, 12, 13, 14, 15);
    }
#pragma unroll
    for (int mi = 0; mi < 4; ++mi)
#pragma unroll
      for (int ni = 0; ni < 2; ++ni)
        acc[mi][ni] = __builtin_amdgcn_wmma_f32_16x16x32_f16(
            false, afrag[mi], false, bfrag[ni], (short)0, acc[mi][ni], false,
            false);
  };

  load_stage(0, 0);
  wait_async_lds();
  __syncthreads();

  for (int kt = 0; kt < NKT; ++kt) {
    int cur = kt & 1;
    if (kt + 1 < NKT) load_stage(cur ^ 1, kt + 1);
    compute_stage(cur);
    wait_async_lds();
    __syncthreads();
  }

#pragma unroll
  for (int mi = 0; mi < 4; ++mi) {
#pragma unroll
    for (int ni = 0; ni < 2; ++ni) {
      int col = n0 + wn + ni * 16 + ml;
      int row0 = m0 + wm + mi * 16 + kh * 8;
      float bv = (float)bias[col];
#pragma unroll
      for (int r = 0; r < 8; ++r)
        out[(size_t)(row0 + r) * NDIM + col] = acc[mi][ni][r] + bv;
    }
  }
}

// ---------------------------------------------------------------------------
// Fallback: fused gather-dequant GEMM (no large workspace needed).
// ---------------------------------------------------------------------------
__global__ __launch_bounds__(256) void exl2_gemm_fused_kernel(
    const _Float16* __restrict__ x, const int* __restrict__ qw,
    const float* __restrict__ scales, const int* __restrict__ invperm,
    const _Float16* __restrict__ bias, float* __restrict__ out) {
  __shared__ __align__(16) _Float16 Ab[2][BM * BK];
  __shared__ __align__(16) _Float16 Bb[2][BN * BK];

  const int tid = threadIdx.x;
  const int m0 = blockIdx.y * BM;
  const int n0 = blockIdx.x * BN;

  const int lane = tid & 31;
  const int wave = tid >> 5;
  const int wm = (wave >> 2) * 64;
  const int wn = (wave & 3) * 32;
  const int ml = lane & 15;
  const int kh = lane >> 4;

  v8f acc[4][2] = {};

  auto load_stage = [&](int st, int kt) {
#pragma unroll
    for (int i = 0; i < 2; ++i) {
      int lin = tid + i * 256;
      int row = lin >> 2;
      int chunk = lin & 3;
      async_copy_b128(x + (size_t)(m0 + row) * KDIM + kt * BK + chunk * 8,
                      &Ab[st][row * BK + chunk * 8]);
    }
    int nloc = tid & 127;
    int kkbase = (tid >> 7) * 16;
    _Float16 tmp[16];
#pragma unroll
    for (int i = 0; i < 16; ++i) {
      int kk = kkbase + i;
      int ks = invperm[kt * BK + kk];
      int w = qw[(size_t)(ks >> 3) * NDIM + n0 + nloc];
      float sc = scales[(size_t)(ks >> 7) * NDIM + n0 + nloc];
      int nib = (w >> ((ks & 7) * 4)) & 0xF;
      tmp[i] = (_Float16)((float)(nib - 8) * sc);
    }
    v8h* dst = (v8h*)&Bb[st][nloc * BK + kkbase];
    dst[0] = *(v8h*)&tmp[0];
    dst[1] = *(v8h*)&tmp[8];
  };

  auto compute_stage = [&](int st) {
    v16h afrag[4], bfrag[2];
#pragma unroll
    for (int mi = 0; mi < 4; ++mi) {
      const _Float16* base = &Ab[st][(wm + mi * 16 + ml) * BK + kh * 8];
      v8h lo = *(const v8h*)base;
      v8h hi = *(const v8h*)(base + 16);
      afrag[mi] = __builtin_shufflevector(lo, hi, 0, 1, 2, 3, 4, 5, 6, 7, 8, 9,
                                          10, 11, 12, 13, 14, 15);
    }
#pragma unroll
    for (int ni = 0; ni < 2; ++ni) {
      const _Float16* base = &Bb[st][(wn + ni * 16 + ml) * BK + kh * 8];
      v8h lo = *(const v8h*)base;
      v8h hi = *(const v8h*)(base + 16);
      bfrag[ni] = __builtin_shufflevector(lo, hi, 0, 1, 2, 3, 4, 5, 6, 7, 8, 9,
                                          10, 11, 12, 13, 14, 15);
    }
#pragma unroll
    for (int mi = 0; mi < 4; ++mi)
#pragma unroll
      for (int ni = 0; ni < 2; ++ni)
        acc[mi][ni] = __builtin_amdgcn_wmma_f32_16x16x32_f16(
            false, afrag[mi], false, bfrag[ni], (short)0, acc[mi][ni], false,
            false);
  };

  load_stage(0, 0);
  wait_async_lds();
  __syncthreads();

  for (int kt = 0; kt < NKT; ++kt) {
    int cur = kt & 1;
    if (kt + 1 < NKT) {
      __builtin_prefetch(invperm + (kt + 1) * BK, 0, 0);
      load_stage(cur ^ 1, kt + 1);
    }
    compute_stage(cur);
    wait_async_lds();
    __syncthreads();
  }

#pragma unroll
  for (int mi = 0; mi < 4; ++mi) {
#pragma unroll
    for (int ni = 0; ni < 2; ++ni) {
      int col = n0 + wn + ni * 16 + ml;
      int row0 = m0 + wm + mi * 16 + kh * 8;
      float bv = (float)bias[col];
#pragma unroll
      for (int r = 0; r < 8; ++r)
        out[(size_t)(row0 + r) * NDIM + col] = acc[mi][ni][r] + bv;
    }
  }
}

// ---------------------------------------------------------------------------
extern "C" void kernel_launch(void* const* d_in, const int* in_sizes, int n_in,
                              void* d_out, int out_size, void* d_ws,
                              size_t ws_size, hipStream_t stream) {
  (void)in_sizes; (void)n_in; (void)out_size;
  const _Float16* input       = (const _Float16*)d_in[0];
  const int*      q_weight    = (const int*)d_in[1];
  const int*      q_scale     = (const int*)d_in[2];
  const _Float16* q_scale_max = (const _Float16*)d_in[3];
  const int*      q_invperm   = (const int*)d_in[4];
  const _Float16* bias        = (const _Float16*)d_in[5];
  float*          out         = (float*)d_out;

  // ws layout: [0, 1.5MB) fp32 scale table | [2MB, 2MB+96MB) W_eff^T fp16
  float* scales = (float*)d_ws;
  const size_t wt_off = 2ull << 20;
  const size_t need = wt_off + (size_t)KDIM * NDIM * sizeof(_Float16);

  int total = GDIM * NDIM;
  exl2_scales_kernel<<<(total + 255) / 256, 256, 0, stream>>>(q_scale,
                                                              q_scale_max,
                                                              scales);

  dim3 grid(NDIM / BN, MDIM / BM);  // 96 x 32 workgroups
  if (ws_size >= need) {
    _Float16* wt = (_Float16*)((char*)d_ws + wt_off);
    int wtotal = NDIM * (KDIM / 8);
    exl2_build_wt_kernel<<<(wtotal + 255) / 256, 256, 0, stream>>>(
        q_weight, scales, q_invperm, wt);
    exl2_gemm_pre_kernel<<<grid, 256, 0, stream>>>(input, wt, bias, out);
  } else {
    exl2_gemm_fused_kernel<<<grid, 256, 0, stream>>>(input, q_weight, scales,
                                                     q_invperm, bias, out);
  }
}
